// AxwinLowMixear_85117661872681
// MI455X (gfx1250) — matrix operand order repaired and checked by reading, not compiled, not run on hardware
//
#include <hip/hip_runtime.h>
#include <hip/hip_bf16.h>

typedef __bf16 bf16_t;
typedef __attribute__((ext_vector_type(16))) __bf16 v16bf;
typedef __attribute__((ext_vector_type(8)))  __bf16 bf16x8;
typedef __attribute__((ext_vector_type(8)))  float  v8f;

#define WMMA_BF16(A_, B_, C_) \
  __builtin_amdgcn_wmma_f32_16x16x32_bf16(false, (A_), false, (B_), (short)0, (C_), false, false)

#define BATCH 4
#define DIM   384
#define NTOK  3136
#define CS_SCALE 0.14433756729740646f   // 48^-0.5
#define DN_SCALE 0.125f                 // 64^-0.5

// ============ WMMA fragment helpers (ISA 7.12.2 layouts, vectorized) ============
__device__ __forceinline__ v16bf load_frag_a(const bf16_t* src, int stride) {
  const int lane = threadIdx.x & 31;
  const int r  = lane & 15;
  const int kh = (lane >> 4) << 3;
  bf16x8 lo = *(const bf16x8*)(src + r * stride + kh);
  bf16x8 hi = *(const bf16x8*)(src + r * stride + kh + 16);
  return __builtin_shufflevector(lo, hi, 0,1,2,3,4,5,6,7,8,9,10,11,12,13,14,15);
}
__device__ __forceinline__ v16bf load_frag_bT(const bf16_t* src, int stride) {
  const int lane = threadIdx.x & 31;
  const int c  = lane & 15;
  const int kh = (lane >> 4) << 3;
  bf16x8 lo = *(const bf16x8*)(src + c * stride + kh);
  bf16x8 hi = *(const bf16x8*)(src + c * stride + kh + 16);
  return __builtin_shufflevector(lo, hi, 0,1,2,3,4,5,6,7,8,9,10,11,12,13,14,15);
}
__device__ __forceinline__ v16bf load_frag_pexp(const float* S, int stride,
                                                const float* rowm, const float* rowil,
                                                int row0, int col0, int ncols) {
  const int lane = threadIdx.x & 31;
  const int r  = lane & 15;
  const int kh = (lane >> 4) << 3;
  const float m  = rowm[row0 + r];
  const float il = rowil ? rowil[row0 + r] : 1.0f;
  const float* base = S + (row0 + r) * stride + col0 + kh;
  v16bf a;
#pragma unroll
  for (int h = 0; h < 2; ++h) {
#pragma unroll
    for (int j = 0; j < 8; ++j) {
      int c = col0 + kh + h * 16 + j;
      float v = 0.f;
      if (c < ncols) v = __expf(base[h * 16 + j] - m) * il;
      a[h * 8 + j] = (bf16_t)v;
    }
  }
  return a;
}

// ---- CDNA5 async global->LDS copy (ASYNCcnt-tracked, no VGPR staging) ----
__device__ __forceinline__ void async_b128(bf16_t* lds_dst, const bf16_t* gsrc) {
  unsigned int l = (unsigned int)(unsigned long long)lds_dst;
  asm volatile("global_load_async_to_lds_b128 %0, %1, off"
               :: "v"(l), "v"(gsrc) : "memory");
}
__device__ __forceinline__ void wait_async0() {
  asm volatile("s_wait_asynccnt 0x0" ::: "memory");
}

// ============ kernel 0: tiled transpose + cvt: xa (B,384,3136) f32 -> XT (B,3136,384) bf16 ============
__global__ __launch_bounds__(256) void k_transpose_cvt(const float* __restrict__ xa,
                                                       bf16_t* __restrict__ XT) {
  const int nb = blockIdx.x, cb = blockIdx.y, b = blockIdx.z;
  __shared__ bf16_t T[64 * 72];
  const int tid = threadIdx.x;
  for (int i = tid; i < 64 * 64; i += 256) {
    int cc = i >> 6, nn = i & 63;
    T[cc * 72 + nn] =
        (bf16_t)xa[((size_t)b * DIM + cb * 64 + cc) * NTOK + nb * 64 + nn];
  }
  __syncthreads();
  for (int i = tid; i < 64 * 8; i += 256) {
    int nn = i >> 3, cg = (i & 7) << 3;
    bf16x8 v;
#pragma unroll
    for (int j = 0; j < 8; ++j) v[j] = T[(cg + j) * 72 + nn];
    *(bf16x8*)(XT + ((size_t)b * NTOK + nb * 64 + nn) * DIM + cb * 64 + cg) = v;
  }
}

// ============ kernel 1: fuse qkv_w (576x192) * proj_w (192x384) -> bf16 ============
__global__ __launch_bounds__(256) void k_fuse_w(const float* __restrict__ qkv_w,
                                                const float* __restrict__ proj_w,
                                                bf16_t* __restrict__ wout) {
  int idx = blockIdx.x * 256 + threadIdx.x;
  if (idx >= 576 * 384) return;
  int o = idx / 384, i = idx % 384;
  float acc = 0.f;
  for (int c = 0; c < 192; ++c) acc += qkv_w[o * 192 + c] * proj_w[c * 384 + i];
  wout[idx] = (bf16_t)acc;
}

// ============ kernel 2: QKV GEMM -> QKt (token-major q,k) + Vc (channel-major v) ============
__global__ __launch_bounds__(256) void k_qkv_gemm(const bf16_t* __restrict__ Wup,
                                                  const bf16_t* __restrict__ Wdn,
                                                  const bf16_t* __restrict__ XT,
                                                  bf16_t* __restrict__ QKt,
                                                  bf16_t* __restrict__ Vc) {
  const int nb   = blockIdx.x;        // 0..48
  const int ob   = blockIdx.y;        // 0..8  (<=5: q/k rows, >=6: v rows)
  const int line = blockIdx.z >> 2;
  const int b    = blockIdx.z & 3;
  const bf16_t* W  = line ? Wdn : Wup;
  const bf16_t* Xt = XT + (size_t)b * NTOK * DIM;
  bf16_t* QKb = QKt + (size_t)blockIdx.z * NTOK * 384;
  bf16_t* Vb  = Vc  + (size_t)blockIdx.z * 192 * NTOK;

  __shared__ bf16_t As[64 * 48];   // [o][k]
  __shared__ bf16_t Bs[64 * 48];   // [n][k]

  const int tid = threadIdx.x;
  const int wave = tid >> 5;
  const int srow = wave >> 1;
  v8f acc[2];
#pragma unroll
  for (int s = 0; s < 2; ++s)
#pragma unroll
    for (int i = 0; i < 8; ++i) acc[s][i] = 0.f;

  for (int kt = 0; kt < 12; ++kt) {
    {
      int r = tid >> 2, kg = (tid & 3) << 3;
      *(bf16x8*)(As + r * 48 + kg) =
          *(const bf16x8*)(W + (ob * 64 + r) * 384 + kt * 32 + kg);
      *(bf16x8*)(Bs + r * 48 + kg) =
          *(const bf16x8*)(Xt + (size_t)(nb * 64 + r) * DIM + kt * 32 + kg);
    }
    __syncthreads();
    v16bf a = load_frag_a(As + srow * 16 * 48, 48);
#pragma unroll
    for (int s = 0; s < 2; ++s) {
      int scol = (wave & 1) * 2 + s;
      v16bf bb = load_frag_bT(Bs + scol * 16 * 48, 48);
      acc[s] = WMMA_BF16(a, bb, acc[s]);
    }
    __syncthreads();
  }
#pragma unroll
  for (int s = 0; s < 2; ++s) {
    int scol = (wave & 1) * 2 + s;
    int lane = tid & 31, col = lane & 15, rb = (lane >> 4) << 3;
    int orow0 = ob * 64 + srow * 16 + rb;          // 8 consecutive qkv channels
    int ocol  = nb * 64 + scol * 16 + col;         // token
    if (orow0 < 384) {
      bf16x8 v;
#pragma unroll
      for (int i = 0; i < 8; ++i) v[i] = (bf16_t)acc[s][i];
      *(bf16x8*)(QKb + (size_t)ocol * 384 + orow0) = v;    // token-major, one b128
    } else {
#pragma unroll
      for (int i = 0; i < 8; ++i)
        Vb[(size_t)(orow0 - 384 + i) * NTOK + ocol] = (bf16_t)acc[s][i];
    }
  }
}

// ============ kernel 3: CSWin attention + LePE ============
__global__ __launch_bounds__(256) void k_cswin(const bf16_t* __restrict__ QKt,
                                               const bf16_t* __restrict__ Vc,
                                               const float* __restrict__ lw0, const float* __restrict__ lb0,
                                               const float* __restrict__ lw1, const float* __restrict__ lb1,
                                               float* __restrict__ out) {
  const int qb     = blockIdx.x;
  const int head   = blockIdx.y & 1;
  const int win    = blockIdx.y >> 1;
  const int branch = blockIdx.z & 1;
  const int b      = blockIdx.z >> 1;
  const int W_ = branch ? 56 : 7;
  const int H_ = branch ? 7 : 56;

  const bf16_t* QKb = QKt + (size_t)b * NTOK * 384;   // line 0
  const int qch = branch * 96 + head * 48;
  const int kch = 192 + qch;
  const bf16_t* Vgc = Vc + ((size_t)b * 192 + branch * 96 + head * 48) * NTOK;

  __shared__ bf16_t Qs[32 * 72];
  __shared__ bf16_t KVu[64 * 424];   // K: [t][d] stride 64; V: [d][t] stride 424
  __shared__ float  Ss[32 * 400];
  __shared__ float  Osm[32 * 56];
  __shared__ float  rowm[32], rowli[32];

  const int tid = threadIdx.x;
  const int wave = tid >> 5;

  // ---- Q block: 8-channel vector chunks ----
  for (int i = tid; i < 32 * 8; i += 256) {
    int r = i >> 3, cg = (i & 7) << 3;
    int tq = qb * 32 + r;
    bf16x8 v;
#pragma unroll
    for (int j = 0; j < 8; ++j) v[j] = (bf16_t)0.f;
    if (tq < 392 && cg < 48) {
      int hh = tq / W_, ww = tq % W_;
      int h = branch ? (win * 7 + hh) : hh;
      int w = branch ? ww : (win * 7 + ww);
      v = *(const bf16x8*)(QKb + (size_t)(h * 56 + w) * 384 + qch + cg);
    }
    *(bf16x8*)(Qs + r * 72 + cg) = v;
  }
  // ---- K window: 8-channel vector chunks, token-major [t][d] ----
  for (int i = tid; i < 416 * 8; i += 256) {
    int t = i >> 3, cg = (i & 7) << 3;
    bf16x8 v;
#pragma unroll
    for (int j = 0; j < 8; ++j) v[j] = (bf16_t)0.f;
    if (t < 392 && cg < 48) {
      int hh = t / W_, ww = t % W_;
      int h = branch ? (win * 7 + hh) : hh;
      int w = branch ? ww : (win * 7 + ww);
      v = *(const bf16x8*)(QKb + (size_t)(h * 56 + w) * 384 + kch + cg);
    }
    *(bf16x8*)(KVu + t * 64 + cg) = v;
  }
  __syncthreads();

  // ---- S = scale * Q K^T ----
  {
    int rt = wave & 1;
    v16bf a0 = load_frag_a(Qs + rt * 16 * 72, 72);
    v16bf a1 = load_frag_a(Qs + rt * 16 * 72 + 32, 72);
    for (int ct = wave >> 1; ct < 25; ct += 4) {
      v8f acc;
#pragma unroll
      for (int i = 0; i < 8; ++i) acc[i] = 0.f;
      acc = WMMA_BF16(a0, load_frag_bT(KVu + ct * 16 * 64, 64), acc);
      acc = WMMA_BF16(a1, load_frag_bT(KVu + ct * 16 * 64 + 32, 64), acc);
      int lane = tid & 31, col = lane & 15, rb = (lane >> 4) << 3;
#pragma unroll
      for (int i = 0; i < 8; ++i)
        Ss[(rt * 16 + rb + i) * 400 + ct * 16 + col] = acc[i] * CS_SCALE;
    }
  }
  __syncthreads();

  // ---- overwrite KVu with V, channel-major [d][t] ----
  for (int i = tid; i < 64 * 424; i += 256) {
    int c = i / 424, t = i - c * 424;
    bf16_t v = (bf16_t)0.f;
    if (t < 392 && c < 48) {
      int hh = t / W_, ww = t % W_;
      int h = branch ? (win * 7 + hh) : hh;
      int w = branch ? ww : (win * 7 + ww);
      v = Vgc[(size_t)c * NTOK + h * 56 + w];
    }
    KVu[c * 424 + t] = v;
  }
  // ---- softmax stats ----
  if (tid < 128) {
    int r = tid >> 2, tr = tid & 3;
    float m = -1e30f;
    for (int c = tr; c < 392; c += 4) m = fmaxf(m, Ss[r * 400 + c]);
    m = fmaxf(m, __shfl_xor(m, 1));
    m = fmaxf(m, __shfl_xor(m, 2));
    float s = 0.f;
    for (int c = tr; c < 392; c += 4) s += __expf(Ss[r * 400 + c] - m);
    s += __shfl_xor(s, 1);
    s += __shfl_xor(s, 2);
    if (tr == 0) { rowm[r] = m; rowli[r] = 1.f / s; }
  }
  __syncthreads();

  // ---- O = P V ----
  if (wave < 6) {
    int rt = wave / 3, ct = wave % 3;
    v8f o;
#pragma unroll
    for (int i = 0; i < 8; ++i) o[i] = 0.f;
    for (int kt = 0; kt < 13; ++kt) {
      v16bf a  = load_frag_pexp(Ss, 400, rowm, rowli, rt * 16, kt * 32, 392);
      v16bf bb = load_frag_bT(KVu + ct * 16 * 424 + kt * 32, 424);
      o = WMMA_BF16(a, bb, o);
    }
    int lane = tid & 31, col = lane & 15, rb = (lane >> 4) << 3;
#pragma unroll
    for (int i = 0; i < 8; ++i)
      Osm[(rt * 16 + rb + i) * 56 + ct * 16 + col] = o[i];
  }
  __syncthreads();

  // ---- LePE + writeout ----
  const float* lw = branch ? lw1 : lw0;
  const float* lb = branch ? lb1 : lb0;
  for (int i = tid; i < 32 * 48; i += 256) {
    int tl = i / 48, cd = i % 48;
    int tq = qb * 32 + tl;
    if (tq >= 392) continue;
    int hh = tq / W_, ww = tq % W_;
    int cch = head * 48 + cd;
    const float* wg = lw + cch * 9;
    float acc = lb[cch];
#pragma unroll
    for (int dy = -1; dy <= 1; ++dy)
#pragma unroll
      for (int dx = -1; dx <= 1; ++dx) {
        int y = hh + dy, x = ww + dx;
        if (y >= 0 && y < H_ && x >= 0 && x < W_)
          acc += wg[(dy + 1) * 3 + dx + 1] * (float)KVu[cd * 424 + y * W_ + x];
      }
    float val = Osm[tl * 56 + cd] + acc;
    int h = branch ? (win * 7 + hh) : hh;
    int w = branch ? ww : (win * 7 + ww);
    out[(((size_t)b * DIM + branch * 96 + cch) * 56 + h) * 56 + w] = val;
  }
}

// ============ kernel 4: global attention, flash-style, async double-buffered K/V ============
__global__ __launch_bounds__(256) void k_global(const bf16_t* __restrict__ QKt,
                                                const bf16_t* __restrict__ Vc,
                                                float* __restrict__ out) {
  const int qb = blockIdx.x, head = blockIdx.y, b = blockIdx.z;
  const bf16_t* T1  = QKt + (size_t)(4 + b) * NTOK * 384;       // line 1
  const bf16_t* Qgl = T1 + head * 64;
  const bf16_t* Kgl = T1 + 192 + head * 64;
  const bf16_t* Vgl = Vc + ((size_t)(4 + b) * 192 + head * 64) * NTOK;

  __shared__ bf16_t Qs[64 * 72];
  __shared__ bf16_t Ks[2][64 * 72];   // [t][d], double-buffered
  __shared__ bf16_t Vt[2][64 * 72];   // [d][t], double-buffered
  __shared__ float  Ss[64 * 72];
  __shared__ float  rm[64], rl[64], ral[64];

  const int tid = threadIdx.x;
  const int wave = tid >> 5;
  const int rt = wave >> 1;

  for (int i = tid; i < 64 * 8; i += 256) {
    int r = i >> 3, cg = (i & 7) << 3;
    *(bf16x8*)(Qs + r * 72 + cg) =
        *(const bf16x8*)(Qgl + (size_t)(qb * 64 + r) * 384 + cg);
  }
  if (tid < 64) { rm[tid] = -1e30f; rl[tid] = 0.f; }

  // issue async DMA for key-tile 0
  for (int i = tid; i < 64 * 8; i += 256) {
    int a = i >> 3, g = (i & 7) << 3;
    async_b128(&Ks[0][a * 72 + g], Kgl + (size_t)(0 * 64 + a) * 384 + g);
    async_b128(&Vt[0][a * 72 + g], Vgl + (size_t)a * NTOK + 0 * 64 + g);
  }
  __syncthreads();

  v16bf qa0 = load_frag_a(Qs + rt * 16 * 72, 72);
  v16bf qa1 = load_frag_a(Qs + rt * 16 * 72 + 32, 72);

  v8f oacc[2];
#pragma unroll
  for (int s = 0; s < 2; ++s)
#pragma unroll
    for (int i = 0; i < 8; ++i) oacc[s][i] = 0.f;

  for (int kt = 0; kt < 49; ++kt) {
    const int cur = kt & 1;
    wait_async0();       // this wave's DMA into buf[cur] done
    __syncthreads();     // all waves' DMA done; prior-iteration readers done
    if (kt < 48) {       // start DMA for next tile into the other buffer
      int nxt = cur ^ 1;
      for (int i = tid; i < 64 * 8; i += 256) {
        int a = i >> 3, g = (i & 7) << 3;
        async_b128(&Ks[nxt][a * 72 + g], Kgl + (size_t)((kt + 1) * 64 + a) * 384 + g);
        async_b128(&Vt[nxt][a * 72 + g], Vgl + (size_t)a * NTOK + (kt + 1) * 64 + g);
      }
    }

    // S tile 64x64
#pragma unroll
    for (int s = 0; s < 2; ++s) {
      int ct = (wave & 1) * 2 + s;
      v8f acc;
#pragma unroll
      for (int i = 0; i < 8; ++i) acc[i] = 0.f;
      acc = WMMA_BF16(qa0, load_frag_bT(Ks[cur] + ct * 16 * 72, 72), acc);
      acc = WMMA_BF16(qa1, load_frag_bT(Ks[cur] + ct * 16 * 72 + 32, 72), acc);
      int lane = tid & 31, col = lane & 15, rb = (lane >> 4) << 3;
#pragma unroll
      for (int i = 0; i < 8; ++i)
        Ss[(rt * 16 + rb + i) * 72 + ct * 16 + col] = acc[i] * DN_SCALE;
    }
    __syncthreads();

    // online softmax stats, 4 threads/row
    {
      int r = tid >> 2, tr = tid & 3;
      float tm = -1e30f;
      for (int c = tr; c < 64; c += 4) tm = fmaxf(tm, Ss[r * 72 + c]);
      tm = fmaxf(tm, __shfl_xor(tm, 1));
      tm = fmaxf(tm, __shfl_xor(tm, 2));
      float mo = rm[r];
      float mn = fmaxf(mo, tm);
      float s = 0.f;
      for (int c = tr; c < 64; c += 4) s += __expf(Ss[r * 72 + c] - mn);
      s += __shfl_xor(s, 1);
      s += __shfl_xor(s, 2);
      if (tr == 0) {
        float al = __expf(mo - mn);
        ral[r] = al;
        rl[r]  = rl[r] * al + s;
        rm[r]  = mn;
      }
    }
    __syncthreads();

    // O := O*alpha + P V
    {
      int lane = tid & 31, rb = (lane >> 4) << 3;
#pragma unroll
      for (int s = 0; s < 2; ++s)
#pragma unroll
        for (int i = 0; i < 8; ++i) oacc[s][i] *= ral[rt * 16 + rb + i];
#pragma unroll
      for (int kk = 0; kk < 2; ++kk) {
        v16bf pa = load_frag_pexp(Ss, 72, rm, nullptr, rt * 16, kk * 32, 64);
#pragma unroll
        for (int s = 0; s < 2; ++s) {
          int ct = (wave & 1) * 2 + s;
          v16bf bb = load_frag_bT(Vt[cur] + ct * 16 * 72 + kk * 32, 72);
          oacc[s] = WMMA_BF16(pa, bb, oacc[s]);
        }
      }
    }
  }
  __syncthreads();
#pragma unroll
  for (int s = 0; s < 2; ++s) {
    int ct = (wave & 1) * 2 + s;
    int lane = tid & 31, col = lane & 15, rb = (lane >> 4) << 3;
#pragma unroll
    for (int i = 0; i < 8; ++i) {
      int r = rt * 16 + rb + i;
      int token = qb * 64 + r;
      int cd = ct * 16 + col;
      float val = oacc[s][i] / rl[r];
      out[(((size_t)b * DIM + 192 + head * 64 + cd) * 56 + token / 56) * 56 + token % 56] = val;
    }
  }
}

// ============ host launcher ============
extern "C" void kernel_launch(void* const* d_in, const int* in_sizes, int n_in,
                              void* d_out, int out_size, void* d_ws, size_t ws_size,
                              hipStream_t stream) {
  const float* xa    = (const float*)d_in[0];
  const float* proj1 = (const float*)d_in[1];
  const float* proj2 = (const float*)d_in[2];
  const float* qkvu  = (const float*)d_in[3];
  const float* qkvd  = (const float*)d_in[4];
  const float* lw0   = (const float*)d_in[5];
  const float* lb0   = (const float*)d_in[6];
  const float* lw1   = (const float*)d_in[7];
  const float* lb1   = (const float*)d_in[8];
  float* out = (float*)d_out;

  char* ws = (char*)d_ws;
  size_t off = 0;
  bf16_t* XT  = (bf16_t*)(ws + off); off += (size_t)BATCH * NTOK * DIM * sizeof(bf16_t);
  bf16_t* Wup = (bf16_t*)(ws + off); off += (size_t)576 * 384 * sizeof(bf16_t);
  bf16_t* Wdn = (bf16_t*)(ws + off); off += (size_t)576 * 384 * sizeof(bf16_t);
  bf16_t* QKt = (bf16_t*)(ws + off); off += (size_t)2 * BATCH * NTOK * 384 * sizeof(bf16_t);
  bf16_t* Vc  = (bf16_t*)(ws + off); off += (size_t)2 * BATCH * 192 * NTOK * sizeof(bf16_t);
  (void)ws_size; (void)in_sizes; (void)n_in; (void)out_size;

  k_transpose_cvt<<<dim3(49, 6, BATCH), 256, 0, stream>>>(xa, XT);
  k_fuse_w<<<(576 * 384 + 255) / 256, 256, 0, stream>>>(qkvu, proj1, Wup);
  k_fuse_w<<<(576 * 384 + 255) / 256, 256, 0, stream>>>(qkvd, proj2, Wdn);
  k_qkv_gemm<<<dim3(49, 9, 8), 256, 0, stream>>>(Wup, Wdn, XT, QKt, Vc);
  k_cswin<<<dim3(13, 16, 8), 256, 0, stream>>>(QKt, Vc, lw0, lb0, lw1, lb1, out);
  k_global<<<dim3(49, 3, 4), 256, 0, stream>>>(QKt, Vc, out);
}
